// EncoderLayer_42399917146737
// MI455X (gfx1250) — matrix operand order repaired and checked by reading, not compiled, not run on hardware
//
#include <hip/hip_runtime.h>
#include <math.h>

typedef __attribute__((ext_vector_type(16))) _Float16 v16h;
typedef __attribute__((ext_vector_type(8)))  float    v8f;

#define B_ 128
#define L_ 1024
#define P_ 512

// ---------------------------------------------------------------------------
// 1) Per-column stats over pc: colcount[l], wbcol[l] = sum_p pc[p,l]*Wb[p],
//    wccol[l] = sum_p pc[p,l]*Wc[p].
// ---------------------------------------------------------------------------
__global__ __launch_bounds__(256) void ecc_colstats(
    const int* __restrict__ pc, const float* __restrict__ Wb,
    const float* __restrict__ Wc, float* __restrict__ colcount,
    float* __restrict__ wbcol, float* __restrict__ wccol) {
  int l = blockIdx.x * blockDim.x + threadIdx.x;
  if (l >= L_) return;
  float cc = 0.f, wb = 0.f, wc = 0.f;
  for (int p = 0; p < P_; ++p) {
    float v = (float)pc[p * L_ + l];          // coalesced across l
    cc += v; wb += v * Wb[p]; wc += v * Wc[p];
  }
  colcount[l] = cc; wbcol[l] = wb; wccol[l] = wc;
}

// ---------------------------------------------------------------------------
// 2) sum(Wr) reduction (512 elements, one block).
// ---------------------------------------------------------------------------
__global__ __launch_bounds__(256) void ecc_sumwr(const float* __restrict__ Wr,
                                                 float* __restrict__ sumWr) {
  __shared__ float red[256];
  int t = threadIdx.x;
  red[t] = Wr[t] + Wr[t + 256];
  __syncthreads();
  for (int s = 128; s > 0; s >>= 1) {
    if (t < s) red[t] += red[t + s];
    __syncthreads();
  }
  if (t == 0) *sumWr = red[0];
}

// ---------------------------------------------------------------------------
// 3) Bit-pack pc rows (P x 32 words) and hard-decision bits (B x 32 words).
// ---------------------------------------------------------------------------
__global__ __launch_bounds__(256) void ecc_packpc(const int* __restrict__ pc,
                                                  unsigned int* __restrict__ pcbits) {
  int t = blockIdx.x * blockDim.x + threadIdx.x;   // 0 .. P_*32-1
  if (t >= P_ * 32) return;
  int p = t >> 5, w = t & 31;
  unsigned int word = 0;
  for (int bit = 0; bit < 32; ++bit)
    word |= (unsigned int)(pc[p * L_ + w * 32 + bit] & 1) << bit;
  pcbits[t] = word;
}

__global__ __launch_bounds__(256) void ecc_packhard(const float* __restrict__ X,
                                                    unsigned int* __restrict__ hardbits) {
  int t = blockIdx.x * blockDim.x + threadIdx.x;   // 0 .. B_*32-1
  if (t >= B_ * 32) return;
  int b = t >> 5, w = t & 31;
  unsigned int word = 0;
  for (int bit = 0; bit < 32; ++bit)
    word |= (unsigned int)(X[b * L_ + w * 32 + bit] < 0.0f) << bit;
  hardbits[t] = word;
}

// ---------------------------------------------------------------------------
// 4) Syndrome h0[b,p] = 1 - 2*parity( AND of bit rows ), stored as f16 (+/-1)
//    for the WMMA matmul; errint[b] |= any parity failure.
// ---------------------------------------------------------------------------
__global__ __launch_bounds__(256) void ecc_h0(const unsigned int* __restrict__ pcbits,
                                              const unsigned int* __restrict__ hardbits,
                                              _Float16* __restrict__ h0f,
                                              int* __restrict__ errint) {
  int t = blockIdx.x * blockDim.x + threadIdx.x;   // 0 .. B_*P_-1
  if (t >= B_ * P_) return;
  int b = t >> 9, p = t & (P_ - 1);
  unsigned int acc = 0;
#pragma unroll
  for (int w = 0; w < 32; ++w) acc ^= pcbits[p * 32 + w] & hardbits[b * 32 + w];
  int par = __popc(acc) & 1;
  h0f[t] = par ? (_Float16)(-1.0f) : (_Float16)(1.0f);
  if (par) atomicOr(&errint[b], 1);
}

// ---------------------------------------------------------------------------
// 5) Bbias = h0 (B_ x P_) @ pc (P_ x L_)  via v_wmma_f32_16x16x32_f16.
//    Exact: h0 in {+-1}, pc in {0,1}, sums <= 512 accumulate exactly in f32.
//    One wave per 16x16 output tile; K loop of 16 WMMAs.
// ---------------------------------------------------------------------------
__global__ __launch_bounds__(32) void ecc_bbias_wmma(
    const _Float16* __restrict__ h0f, const int* __restrict__ pc,
    float* __restrict__ Bbias) {
  const int lane = threadIdx.x;           // 0..31
  const int m0 = blockIdx.x * 16;         // batch tile
  const int n0 = blockIdx.y * 16;         // column tile
  const int half = lane >> 4;             // lane group 0/1
  const int ml = lane & 15;

  v8f c = {};
  for (int k0 = 0; k0 < P_; k0 += 32) {
    v16h a, b;
    // A tile (16x32 f16): lane ml is row M; VGPR j holds K pair per ISA layout.
#pragma unroll
    for (int j = 0; j < 8; ++j) {
      int kb = 2 * j + (j >= 4 ? 8 : 0) + 8 * half;   // ISA 16-bit A 16x32 layout
      const _Float16* row = h0f + (size_t)(m0 + ml) * P_ + k0;
      a[2 * j]     = row[kb];
      a[2 * j + 1] = row[kb + 1];
    }
    // B tile (32x16 f16): lane ml is column N; K order matches A
    // (lanes 0-15: K 0..7,16..23; lanes 16-31: K 8..15,24..31).
#pragma unroll
    for (int j = 0; j < 8; ++j) {
      int k = 2 * j + (j >= 4 ? 8 : 0) + 8 * half;
      b[2 * j]     = (_Float16)(float)pc[(size_t)(k0 + k)     * L_ + n0 + ml];
      b[2 * j + 1] = (_Float16)(float)pc[(size_t)(k0 + k + 1) * L_ + n0 + ml];
    }
    c = __builtin_amdgcn_wmma_f32_16x16x32_f16(
        /*neg_a=*/false, a, /*neg_b=*/false, b,
        /*c_mod=*/(short)0, c, /*reuse_a=*/false, /*reuse_b=*/false);
  }
  // C/D layout: lane ml = N, VGPR r -> row r + 8*half.
#pragma unroll
  for (int r = 0; r < 8; ++r)
    Bbias[(size_t)(m0 + r + 8 * half) * L_ + n0 + ml] = c[r];
}

// ---------------------------------------------------------------------------
// 6) Global max|Bbias| via monotone uint bits + atomicMax.
// ---------------------------------------------------------------------------
__global__ __launch_bounds__(256) void ecc_absmax(const float* __restrict__ Bbias,
                                                  unsigned int* __restrict__ maxbits,
                                                  int n) {
  __shared__ unsigned int red[256];
  int t = threadIdx.x;
  unsigned int v = 0;
  for (int i = blockIdx.x * blockDim.x + t; i < n; i += gridDim.x * blockDim.x)
    v = max(v, __float_as_uint(fabsf(Bbias[i])));
  red[t] = v;
  __syncthreads();
  for (int s = 128; s > 0; s >>= 1) {
    if (t < s) red[t] = max(red[t], red[t + s]);
    __syncthreads();
  }
  if (t == 0) atomicMax(maxbits, red[0]);
}

// ---------------------------------------------------------------------------
// 7) Final: v[b,l] = 2*err[b]*C*Bcoef*colcount*sumWr + br, softmax over L.
//    One workgroup per batch row.
// ---------------------------------------------------------------------------
__global__ __launch_bounds__(256) void ecc_final_softmax(
    const float* __restrict__ X, const float* __restrict__ Bbias,
    const float* __restrict__ colcount, const float* __restrict__ wbcol,
    const float* __restrict__ wccol, const int* __restrict__ errint,
    const unsigned int* __restrict__ maxbits, const float* __restrict__ sumWr,
    const float* __restrict__ br, float* __restrict__ out) {
  __shared__ float vals[L_];
  __shared__ float red[256];
  const int b = blockIdx.x, t = threadIdx.x;
  const float mabs = __uint_as_float(*maxbits);
  const float inv_mabs = 1.0f / mabs;            // matches reference (nan/inf if 0)
  const float sw = *sumWr;
  const float bias = *br;
  const float err = (errint[b] != 0) ? 1.0f : 0.0f;

  for (int l = t; l < L_; l += 256) {
    float x  = X[(size_t)b * L_ + l];
    float ax = fabsf(x);
    float bc = Bbias[(size_t)b * L_ + l] * inv_mabs + tanhf(ax * wbcol[l]);
    float cc = 0.5f + tanhf(ax * wccol[l]);
    vals[l] = 2.0f * err * cc * bc * colcount[l] * sw + bias;
  }
  __syncthreads();

  float m = -INFINITY;
  for (int l = t; l < L_; l += 256) m = fmaxf(m, vals[l]);
  red[t] = m;
  __syncthreads();
  for (int s = 128; s > 0; s >>= 1) {
    if (t < s) red[t] = fmaxf(red[t], red[t + s]);
    __syncthreads();
  }
  const float gmax = red[0];
  __syncthreads();

  float ssum = 0.f;
  for (int l = t; l < L_; l += 256) {
    float e = expf(vals[l] - gmax);
    vals[l] = e;
    ssum += e;
  }
  red[t] = ssum;
  __syncthreads();
  for (int s = 128; s > 0; s >>= 1) {
    if (t < s) red[t] += red[t + s];
    __syncthreads();
  }
  const float rinv = 1.0f / red[0];
  __syncthreads();
  for (int l = t; l < L_; l += 256) out[(size_t)b * L_ + l] = vals[l] * rinv;
}

// ---------------------------------------------------------------------------
// Host-side launch
// ---------------------------------------------------------------------------
extern "C" void kernel_launch(void* const* d_in, const int* in_sizes, int n_in,
                              void* d_out, int out_size, void* d_ws, size_t ws_size,
                              hipStream_t stream) {
  const float* X  = (const float*)d_in[0];   // (B,L,1)
  const int*   pc = (const int*)d_in[1];     // (P,L)
  const float* Wb = (const float*)d_in[2];   // (1,P)
  const float* Wc = (const float*)d_in[3];
  const float* Wr = (const float*)d_in[4];
  const float* br = (const float*)d_in[5];   // (1,)
  float* out = (float*)d_out;                // (B,L,1)
  (void)in_sizes; (void)n_in; (void)out_size; (void)ws_size;

  // Workspace layout (bytes, 256-aligned chunks).
  char* w = (char*)d_ws;
  float*        colcount = (float*)(w + 0);          //   4096
  float*        wbcol    = (float*)(w + 4096);       //   4096
  float*        wccol    = (float*)(w + 8192);       //   4096
  unsigned int* pcbits   = (unsigned int*)(w + 12288);   // 65536
  unsigned int* hardbits = (unsigned int*)(w + 77824);   // 16384
  _Float16*     h0f      = (_Float16*)(w + 94208);       // 131072
  int*          errint   = (int*)(w + 225280);           //   512
  unsigned int* maxbits  = (unsigned int*)(w + 225792);  //     4
  float*        sumWr    = (float*)(w + 225796);         //     4
  float*        Bbias    = (float*)(w + 226048);         // 524288

  // Zero the atomic accumulators each call (graph-capture safe).
  hipMemsetAsync(errint, 0, B_ * sizeof(int), stream);
  hipMemsetAsync(maxbits, 0, sizeof(unsigned int), stream);

  ecc_colstats<<<dim3(L_ / 256), dim3(256), 0, stream>>>(pc, Wb, Wc, colcount, wbcol, wccol);
  ecc_sumwr<<<dim3(1), dim3(256), 0, stream>>>(Wr, sumWr);
  ecc_packpc<<<dim3((P_ * 32) / 256), dim3(256), 0, stream>>>(pc, pcbits);
  ecc_packhard<<<dim3((B_ * 32) / 256), dim3(256), 0, stream>>>(X, hardbits);
  ecc_h0<<<dim3((B_ * P_) / 256), dim3(256), 0, stream>>>(pcbits, hardbits, h0f, errint);
  ecc_bbias_wmma<<<dim3(B_ / 16, L_ / 16), dim3(32), 0, stream>>>(h0f, pc, Bbias);
  ecc_absmax<<<dim3(128), dim3(256), 0, stream>>>(Bbias, maxbits, B_ * L_);
  ecc_final_softmax<<<dim3(B_), dim3(256), 0, stream>>>(
      X, Bbias, colcount, wbcol, wccol, errint, maxbits, sumWr, br, out);
}
